// WarlightPolicyNetResidual_87995289960633
// MI455X (gfx1250) — matrix operand order repaired and checked by reading, not compiled, not run on hardware
//
#include <hip/hip_runtime.h>

// ============================================================================
// WarlightPolicyNetResidual for MI455X (gfx1250, wave32, WMMA f16->f32)
//
// Input index map (top-level dict insertion order; nested params flattened
// with sorted keys, jax-pytree style):
//  0  x (N,128)
//  1..10  army: l1.W(128,128) l1.b(128) l2.W(128,64) l2.b(64) l3.W(64,50)
//               l3.b(50) n1.b(128) n1.g(128) n2.b(64) n2.g(64)
// 11..20  edge: l1.W(128,64) l1.b(64) l2.W(64,32) l2.b(32) l3.W(32,1)
//               l3.b(1) n1.b(64) n1.g(64) n2.b(32) n2.g(32)
// 21..24  gcn0: lin.W(128,64) lin.b(64) ln.b(64) ln.g(64)
// 25..28  gcn1: lin.W(64,64)  lin.b    ln.b     ln.g
// 29..32  gcn2: lin.W(64,64)  lin.b    ln.b     ln.g
// 33..42  place: l1.W(64,64) l1.b(64) l2.W(64,32) l2.b(32) l3.W(32,1)
//                l3.b(1) n1.b(64) n1.g(64) n2.b(32) n2.g(32)
// 43 proj0 (128,64)   44 edge_index (2,E)   45 action_edges (A,2)
// 46 army_counts (N)
//
// Output: [placement(N) | edge_logits(A) | army_logits(A*50)] fp32
//
// LDS data is stored in WMMA-fragment-ready layout so each fragment load is
// one contiguous 32B (v16h) read per lane: 2x ds_load_b128 vs 16x ds_load_u16.
// ============================================================================

typedef __attribute__((ext_vector_type(16))) _Float16 v16h;
typedef __attribute__((ext_vector_type(8)))  float    v8f;

#define LN_EPS 1e-5f

__device__ __forceinline__ v8f wmma16(v16h a, v16h b, v8f c) {
  return __builtin_amdgcn_wmma_f32_16x16x32_f16(false, a, false, b,
                                                (short)0, c, false, false);
}

// ---- fragment-layout swizzles ----------------------------------------------
// B matrix (K x CO): fragment (kc,nt) occupies a 32-lane x 16-half block.
// Lane (h<<4)+(n&15) holds halves j: k = kc*32 + {h*8+j (j<8), 16+h*8+j-8 (j>=8)}.
__device__ __forceinline__ int b_off(int k, int n, int CO16) {
  int kc = k >> 5;
  int h  = (k >> 3) & 1;
  int j  = (k & 7) + (((k >> 4) & 1) << 3);
  int nt = n >> 4;
  int lane = (h << 4) + (n & 15);
  return ((kc * CO16 + nt) * 32 + lane) * 16 + j;
}
// A tile (16 rows x K): fragment kc occupies a 32-lane x 16-half block.
// Lane m+(h<<4) holds halves j: k = kc*32 + {h*8+j (j<8), 16+h*8+j-8 (j>=8)}.
__device__ __forceinline__ int a_off(int m, int k) {
  int kc   = k >> 5;
  int krem = k & 31;
  int h    = (krem >> 3) & 1;                    // [8,16) or [24,32) -> hi
  int j    = (krem & 7) + ((krem >> 4) << 3);    // +8 for krem >= 16
  int lane = m + (h << 4);
  return (kc * 32 + lane) * 16 + j;
}

// Single contiguous fragment loads from swizzled LDS.
__device__ __forceinline__ v16h load_b_frag(const _Float16* w, int kc, int nt,
                                            int CO16, int lane) {
  return ((const v16h*)w)[(kc * CO16 + nt) * 32 + lane];
}
__device__ __forceinline__ v16h load_a_frag(const _Float16* tile, int kc,
                                            int lane) {
  return ((const v16h*)tile)[kc * 32 + lane];
}

// A fragment (16x32 f16) from a global f32 row, optional symmetric clip.
__device__ __forceinline__ v16h load_a_f32(const float* __restrict__ row,
                                           int kbase, int lane, float clipv) {
  int kb = (lane >> 4) * 8;
  v16h a;
#pragma unroll
  for (int j = 0; j < 8; ++j) {
    float x0 = row[kbase + kb + j];
    float x1 = row[kbase + 16 + kb + j];
    if (clipv > 0.f) {
      x0 = fminf(fmaxf(x0, -clipv), clipv);
      x1 = fminf(fmaxf(x1, -clipv), clipv);
    }
    a[j]     = (_Float16)x0;
    a[j + 8] = (_Float16)x1;
  }
  return a;
}

// Row-wise LayerNorm + ReLU from f32 LDS tile (row-major) into f16 LDS tile
// stored in A-fragment layout (ready for the next layer's WMMA).
__device__ __forceinline__ void ln_relu_f16(const float* __restrict__ scr,
                                            _Float16* __restrict__ dst,
                                            const float* __restrict__ g,
                                            const float* __restrict__ b,
                                            int CO, int lane) {
  if (lane < 16) {
    const float* rp = scr + lane * CO;
    float s = 0.f, ss = 0.f;
    for (int j = 0; j < CO; ++j) { float v = rp[j]; s += v; ss += v * v; }
    float mu  = s / (float)CO;
    float var = ss / (float)CO - mu * mu;
    float inv = rsqrtf(var + LN_EPS);
    for (int j = 0; j < CO; ++j) {
      float v = (rp[j] - mu) * inv * g[j] + b[j];
      dst[a_off(lane, j)] = (_Float16)(v > 0.f ? v : 0.f);
    }
  }
}

// ---------------------------------------------------------------------------
// Degree / normalization kernels
// ---------------------------------------------------------------------------
__global__ void init_deg_kernel(float* deg, int N) {
  int i = blockIdx.x * blockDim.x + threadIdx.x;
  if (i < N) deg[i] = 1.0f;
}
__global__ void accum_deg_kernel(const int* __restrict__ tgt, float* deg, int E) {
  int i = blockIdx.x * blockDim.x + threadIdx.x;
  if (i < E) atomicAdd(&deg[tgt[i]], 1.0f);
}
__global__ void rsqrt_deg_kernel(const float* __restrict__ deg, float* dis, int N) {
  int i = blockIdx.x * blockDim.x + threadIdx.x;
  if (i < N) dis[i] = rsqrtf(deg[i]);
}

// ---------------------------------------------------------------------------
// Generic WMMA GEMM: C[M,CO] = A[M,K] @ W[K,CO]   (fp32 in/out, f16 compute)
// 128 threads = 4 waves; each wave owns a 16-row tile; weights staged in LDS
// pre-swizzled into B-fragment layout.
// ---------------------------------------------------------------------------
template <int K, int CO>
__global__ void __launch_bounds__(128) gemm_wmma_kernel(
    const float* __restrict__ Ain, const float* __restrict__ W,
    float* __restrict__ Cout, int M) {
  __shared__ alignas(32) _Float16 wl[K * CO];
  int tid = threadIdx.x;
  for (int i = tid; i < K * CO; i += 128) {
    int k = i / CO, n = i % CO;
    wl[b_off(k, n, CO / 16)] = (_Float16)W[i];
  }
  __syncthreads();

  int wid = tid >> 5, lane = tid & 31;
  int m0 = (blockIdx.x * 4 + wid) * 16;
  int row = m0 + (lane & 15);
  if (row >= M) row = M - 1;
  const float* arow = Ain + (size_t)row * K;

  v16h af[K / 32];
#pragma unroll
  for (int kc = 0; kc < K / 32; ++kc) af[kc] = load_a_f32(arow, kc * 32, lane, 0.f);

#pragma unroll
  for (int nt = 0; nt < CO / 16; ++nt) {
    v8f acc = {};
#pragma unroll
    for (int kc = 0; kc < K / 32; ++kc)
      acc = wmma16(af[kc], load_b_frag(wl, kc, nt, CO / 16, lane), acc);
    int n = (lane & 15) + nt * 16;
#pragma unroll
    for (int r = 0; r < 8; ++r) {
      int m = m0 + r + ((lane >> 4) << 3);
      if (m < M) Cout[(size_t)m * CO + n] = acc[r];
    }
  }
}

// ---------------------------------------------------------------------------
// GCN aggregate: agg = xw/deg + bias, then scatter += norm_e * xw[src]
// ---------------------------------------------------------------------------
__global__ void init_agg_kernel(const float* __restrict__ xw,
                                const float* __restrict__ deg,
                                const float* __restrict__ bias,
                                float* __restrict__ agg, int N) {
  int idx = blockIdx.x * blockDim.x + threadIdx.x;
  if (idx < N * 64) {
    int n = idx >> 6, d = idx & 63;
    agg[idx] = xw[idx] / deg[n] + bias[d];
  }
}

__global__ void scatter_kernel(const int* __restrict__ src,
                               const int* __restrict__ tgt,
                               const float* __restrict__ dis,
                               const float* __restrict__ xw,
                               float* __restrict__ agg, int E) {
  int idx = blockIdx.x * blockDim.x + threadIdx.x;
  if (idx >= E * 16) return;
  int e = idx >> 4;
  int c = (idx & 15) << 2;  // 4 floats per thread
  int s = src[e], t = tgt[e];
  float nm = dis[s] * dis[t];
  const float4 v = *(const float4*)(xw + (size_t)s * 64 + c);
  float* ap = agg + (size_t)t * 64 + c;
  atomicAdd(ap + 0, nm * v.x);
  atomicAdd(ap + 1, nm * v.y);
  atomicAdd(ap + 2, nm * v.z);
  atomicAdd(ap + 3, nm * v.w);
}

// LN(agg)*g+b + ident, relu -> out.  One wave32 per node (2 elems/lane).
__global__ void __launch_bounds__(256) gcn_post_kernel(
    const float* __restrict__ agg, const float* __restrict__ ident,
    const float* __restrict__ g, const float* __restrict__ b,
    float* __restrict__ out, int N) {
  int node = blockIdx.x * 8 + (threadIdx.x >> 5);
  int lane = threadIdx.x & 31;
  if (node >= N) return;
  size_t base = (size_t)node * 64;
  float v0 = agg[base + lane], v1 = agg[base + lane + 32];
  float s = v0 + v1;
  for (int off = 16; off; off >>= 1) s += __shfl_xor(s, off, 32);
  float mu = s * (1.f / 64.f);
  float d0 = v0 - mu, d1 = v1 - mu;
  float ss = d0 * d0 + d1 * d1;
  for (int off = 16; off; off >>= 1) ss += __shfl_xor(ss, off, 32);
  float inv = rsqrtf(ss * (1.f / 64.f) + LN_EPS);
  float o0 = d0 * inv * g[lane] + b[lane] + ident[base + lane];
  float o1 = d1 * inv * g[lane + 32] + b[lane + 32] + ident[base + lane + 32];
  out[base + lane]      = o0 > 0.f ? o0 : 0.f;
  out[base + lane + 32] = o1 > 0.f ? o1 : 0.f;
}

// ---------------------------------------------------------------------------
// Fused placement head: emb(64) -> 64 -> 32 -> 1, LN+ReLU between, clip +-20.
// 4 waves/block, each wave = 16 nodes, weights in LDS (fragment layout).
// ---------------------------------------------------------------------------
__global__ void __launch_bounds__(128) place_mlp_kernel(
    const float* __restrict__ emb,
    const float* __restrict__ w1g, const float* __restrict__ b1g,
    const float* __restrict__ g1g, const float* __restrict__ bb1g,
    const float* __restrict__ w2g, const float* __restrict__ b2g,
    const float* __restrict__ g2g, const float* __restrict__ bb2g,
    const float* __restrict__ w3g, const float* __restrict__ b3g,
    float* __restrict__ out, int M) {
  __shared__ alignas(32) _Float16 W1[64 * 64];
  __shared__ alignas(32) _Float16 W2[64 * 32];
  __shared__ float P[384];  // b1@0 g1@64 bb1@128 b2@192 g2@224 bb2@256 w3@288 b3@320
  __shared__ float scrF[4][16 * 64];
  __shared__ alignas(32) _Float16 scrH[4][16 * 64];

  int tid = threadIdx.x;
  for (int i = tid; i < 64 * 64; i += 128)
    W1[b_off(i >> 6, i & 63, 4)] = (_Float16)w1g[i];
  for (int i = tid; i < 64 * 32; i += 128)
    W2[b_off(i >> 5, i & 31, 2)] = (_Float16)w2g[i];
  for (int i = tid; i < 64; i += 128) {
    P[i] = b1g[i]; P[64 + i] = g1g[i]; P[128 + i] = bb1g[i];
  }
  for (int i = tid; i < 32; i += 128) {
    P[192 + i] = b2g[i]; P[224 + i] = g2g[i]; P[256 + i] = bb2g[i];
    P[288 + i] = w3g[i];
  }
  if (tid == 0) P[320] = b3g[0];
  __syncthreads();

  int wid = tid >> 5, lane = tid & 31;
  int m0 = (blockIdx.x * 4 + wid) * 16;
  int row = m0 + (lane & 15);
  if (row >= M) row = M - 1;
  const float* arow = emb + (size_t)row * 64;

  // l1: 64 -> 64
  v16h a0 = load_a_f32(arow, 0, lane, 0.f);
  v16h a1 = load_a_f32(arow, 32, lane, 0.f);
#pragma unroll
  for (int nt = 0; nt < 4; ++nt) {
    v8f acc = {};
    acc = wmma16(a0, load_b_frag(W1, 0, nt, 4, lane), acc);
    acc = wmma16(a1, load_b_frag(W1, 1, nt, 4, lane), acc);
    int n = (lane & 15) + nt * 16;
#pragma unroll
    for (int r = 0; r < 8; ++r)
      scrF[wid][(r + ((lane >> 4) << 3)) * 64 + n] = acc[r] + P[n];
  }
  __syncthreads();
  ln_relu_f16(scrF[wid], scrH[wid], P + 64, P + 128, 64, lane);
  __syncthreads();

  // l2: 64 -> 32
  v16h h0 = load_a_frag(scrH[wid], 0, lane);
  v16h h1 = load_a_frag(scrH[wid], 1, lane);
#pragma unroll
  for (int nt = 0; nt < 2; ++nt) {
    v8f acc = {};
    acc = wmma16(h0, load_b_frag(W2, 0, nt, 2, lane), acc);
    acc = wmma16(h1, load_b_frag(W2, 1, nt, 2, lane), acc);
    int n = (lane & 15) + nt * 16;
#pragma unroll
    for (int r = 0; r < 8; ++r)
      scrF[wid][(r + ((lane >> 4) << 3)) * 32 + n] = acc[r] + P[192 + n];
  }
  __syncthreads();
  ln_relu_f16(scrF[wid], scrH[wid], P + 224, P + 256, 32, lane);
  __syncthreads();

  // l3: 32 -> 1, clip
  if (lane < 16) {
    float v = P[320];
    for (int k = 0; k < 32; ++k)
      v += (float)scrH[wid][a_off(lane, k)] * P[288 + k];
    v = fminf(fmaxf(v, -20.f), 20.f);
    int node = m0 + lane;
    if (node < M) out[node] = v;
  }
}

// ---------------------------------------------------------------------------
// Fused edge + army heads.  Per 16 action rows: gather+clip concat embedding
// once into A fragments (kept in registers across both heads), run edge MLP
// (128->64->32->1) and army MLP (128->128->64->50 padded to 64) with all
// weights resident in LDS (f16, fragment layout).
// ---------------------------------------------------------------------------
__global__ void __launch_bounds__(128) edge_army_kernel(
    const float* __restrict__ emb,
    const int* __restrict__ ae,      // (A,2)
    const int* __restrict__ army,    // (N,)
    const float* eW1, const float* eB1, const float* eG1, const float* eBB1,
    const float* eW2, const float* eB2, const float* eG2, const float* eBB2,
    const float* eW3, const float* eB3,
    const float* aW1, const float* aB1, const float* aG1, const float* aBB1,
    const float* aW2, const float* aB2, const float* aG2, const float* aBB2,
    const float* aW3, const float* aB3,
    float* __restrict__ outE, float* __restrict__ outA,
    int A, int N) {
  __shared__ alignas(32) _Float16 sa1[128 * 128];  // army l1
  __shared__ alignas(32) _Float16 sa2[128 * 64];   // army l2
  __shared__ alignas(32) _Float16 sa3[64 * 64];    // army l3 (padded 50->64)
  __shared__ alignas(32) _Float16 se1[128 * 64];   // edge l1
  __shared__ alignas(32) _Float16 se2[64 * 32];    // edge l2
  // eb1@0 eg1@64 ebb1@128 eb2@192 eg2@224 ebb2@256 ew3@288
  // ab1@320 ag1@448 abb1@576 ab2@704 ag2@768 abb2@832 ab3@896 eb3@960
  __shared__ float prm[1024];
  __shared__ float scrF[4][16 * 128];
  __shared__ alignas(32) _Float16 scrH[4][16 * 128];
  __shared__ int   scN[4][16], tcN[4][16], arS[4][16];
  __shared__ float pen[4][16];

  int tid = threadIdx.x;
  for (int i = tid; i < 128 * 128; i += 128)
    sa1[b_off(i >> 7, i & 127, 8)] = (_Float16)aW1[i];
  for (int i = tid; i < 128 * 64; i += 128)
    sa2[b_off(i >> 6, i & 63, 4)] = (_Float16)aW2[i];
  for (int i = tid; i < 64 * 64; i += 128) {
    int k = i >> 6, n = i & 63;
    sa3[b_off(k, n, 4)] = (n < 50) ? (_Float16)aW3[k * 50 + n] : (_Float16)0.f;
  }
  for (int i = tid; i < 128 * 64; i += 128)
    se1[b_off(i >> 6, i & 63, 4)] = (_Float16)eW1[i];
  for (int i = tid; i < 64 * 32; i += 128)
    se2[b_off(i >> 5, i & 31, 2)] = (_Float16)eW2[i];
  for (int i = tid; i < 64; i += 128) {
    prm[i] = eB1[i]; prm[64 + i] = eG1[i]; prm[128 + i] = eBB1[i];
    prm[704 + i] = aB2[i]; prm[768 + i] = aG2[i]; prm[832 + i] = aBB2[i];
    prm[896 + i] = (i < 50) ? aB3[i] : 0.f;
  }
  for (int i = tid; i < 32; i += 128) {
    prm[192 + i] = eB2[i]; prm[224 + i] = eG2[i]; prm[256 + i] = eBB2[i];
    prm[288 + i] = eW3[i];
  }
  for (int i = tid; i < 128; i += 128) {
    prm[320 + i] = aB1[i]; prm[448 + i] = aG1[i]; prm[576 + i] = aBB1[i];
  }
  if (tid == 0) prm[960] = eB3[0];
  __syncthreads();

  int wid = tid >> 5, lane = tid & 31;
  int i0 = (blockIdx.x * 4 + wid) * 16;

  // Per-row metadata (clipped node ids, penalty, army counts)
  if (lane < 16) {
    int i  = i0 + lane;
    int iq = (i < A) ? i : (A - 1);
    int as = ae[2 * iq], at = ae[2 * iq + 1];
    int sc = as < 0 ? 0 : (as > N - 1 ? N - 1 : as);
    int tc = at < 0 ? 0 : (at > N - 1 ? N - 1 : at);
    int sa = army[sc], ta = army[tc];
    bool valid = (as >= 0) && (at >= 0);
    bool bad = valid && ((sa <= 2) || (ta >= 3 * sa));
    float p = (bad ? 1.f : 0.f) + ((valid && as == at) ? 100.f : 0.f);
    scN[wid][lane] = sc; tcN[wid][lane] = tc;
    arS[wid][lane] = sa; pen[wid][lane] = p;
  }
  __syncthreads();

  // Shared input fragments: concat(emb[sc], emb[tc]) clipped to +-10, f16
  v16h aIn[4];
  {
    int r = lane & 15;
    const float* rs = emb + (size_t)scN[wid][r] * 64;
    const float* rt = emb + (size_t)tcN[wid][r] * 64;
    aIn[0] = load_a_f32(rs, 0, lane, 10.f);
    aIn[1] = load_a_f32(rs, 32, lane, 10.f);
    aIn[2] = load_a_f32(rt, 0, lane, 10.f);
    aIn[3] = load_a_f32(rt, 32, lane, 10.f);
  }

  // ---------------- edge head: 128 -> 64 -> 32 -> 1 ----------------
#pragma unroll
  for (int nt = 0; nt < 4; ++nt) {
    v8f acc = {};
#pragma unroll
    for (int kc = 0; kc < 4; ++kc)
      acc = wmma16(aIn[kc], load_b_frag(se1, kc, nt, 4, lane), acc);
    int n = (lane & 15) + nt * 16;
#pragma unroll
    for (int r = 0; r < 8; ++r)
      scrF[wid][(r + ((lane >> 4) << 3)) * 64 + n] = acc[r] + prm[n];
  }
  __syncthreads();
  ln_relu_f16(scrF[wid], scrH[wid], prm + 64, prm + 128, 64, lane);
  __syncthreads();
  {
    v16h h0 = load_a_frag(scrH[wid], 0, lane);
    v16h h1 = load_a_frag(scrH[wid], 1, lane);
#pragma unroll
    for (int nt = 0; nt < 2; ++nt) {
      v8f acc = {};
      acc = wmma16(h0, load_b_frag(se2, 0, nt, 2, lane), acc);
      acc = wmma16(h1, load_b_frag(se2, 1, nt, 2, lane), acc);
      int n = (lane & 15) + nt * 16;
#pragma unroll
      for (int r = 0; r < 8; ++r)
        scrF[wid][(r + ((lane >> 4) << 3)) * 32 + n] = acc[r] + prm[192 + n];
    }
  }
  __syncthreads();
  ln_relu_f16(scrF[wid], scrH[wid], prm + 224, prm + 256, 32, lane);
  __syncthreads();
  if (lane < 16) {
    float v = prm[960];
    for (int k = 0; k < 32; ++k)
      v += (float)scrH[wid][a_off(lane, k)] * prm[288 + k];
    v -= pen[wid][lane];
    v = fminf(fmaxf(v, -20.f), 20.f);
    int i = i0 + lane;
    if (i < A) outE[i] = v;
  }
  __syncthreads();

  // ---------------- army head: 128 -> 128 -> 64 -> 50 ----------------
#pragma unroll
  for (int nt = 0; nt < 8; ++nt) {
    v8f acc = {};
#pragma unroll
    for (int kc = 0; kc < 4; ++kc)
      acc = wmma16(aIn[kc], load_b_frag(sa1, kc, nt, 8, lane), acc);
    int n = (lane & 15) + nt * 16;
#pragma unroll
    for (int r = 0; r < 8; ++r)
      scrF[wid][(r + ((lane >> 4) << 3)) * 128 + n] = acc[r] + prm[320 + n];
  }
  __syncthreads();
  ln_relu_f16(scrF[wid], scrH[wid], prm + 448, prm + 576, 128, lane);
  __syncthreads();
  {
    v16h hf[4];
#pragma unroll
    for (int kc = 0; kc < 4; ++kc) hf[kc] = load_a_frag(scrH[wid], kc, lane);
#pragma unroll
    for (int nt = 0; nt < 4; ++nt) {
      v8f acc = {};
#pragma unroll
      for (int kc = 0; kc < 4; ++kc)
        acc = wmma16(hf[kc], load_b_frag(sa2, kc, nt, 4, lane), acc);
      int n = (lane & 15) + nt * 16;
#pragma unroll
      for (int r = 0; r < 8; ++r)
        scrF[wid][(r + ((lane >> 4) << 3)) * 64 + n] = acc[r] + prm[704 + n];
    }
  }
  __syncthreads();
  ln_relu_f16(scrF[wid], scrH[wid], prm + 768, prm + 832, 64, lane);
  __syncthreads();
  {
    v16h h0 = load_a_frag(scrH[wid], 0, lane);
    v16h h1 = load_a_frag(scrH[wid], 1, lane);
#pragma unroll
    for (int nt = 0; nt < 4; ++nt) {
      v8f acc = {};
      acc = wmma16(h0, load_b_frag(sa3, 0, nt, 4, lane), acc);
      acc = wmma16(h1, load_b_frag(sa3, 1, nt, 4, lane), acc);
      int n = (lane & 15) + nt * 16;
#pragma unroll
      for (int r = 0; r < 8; ++r) {
        int m = r + ((lane >> 4) << 3);
        int i = i0 + m;
        if (i < A && n < 50) {
          float v = acc[r] + prm[896 + n];
          v = fminf(fmaxf(v, -20.f), 20.f);
          if (n >= arS[wid][m]) v = -20.f;   // masked -> -1e9 -> clip -> -20
          outA[(size_t)i * 50 + n] = v;
        }
      }
    }
  }
}

// ---------------------------------------------------------------------------
// Host launcher
// ---------------------------------------------------------------------------
extern "C" void kernel_launch(void* const* d_in, const int* in_sizes, int n_in,
                              void* d_out, int out_size, void* d_ws, size_t ws_size,
                              hipStream_t stream) {
  const float* x      = (const float*)d_in[0];
  const float* a_l1W  = (const float*)d_in[1];
  const float* a_l1b  = (const float*)d_in[2];
  const float* a_l2W  = (const float*)d_in[3];
  const float* a_l2b  = (const float*)d_in[4];
  const float* a_l3W  = (const float*)d_in[5];
  const float* a_l3b  = (const float*)d_in[6];
  const float* a_n1b  = (const float*)d_in[7];
  const float* a_n1g  = (const float*)d_in[8];
  const float* a_n2b  = (const float*)d_in[9];
  const float* a_n2g  = (const float*)d_in[10];
  const float* e_l1W  = (const float*)d_in[11];
  const float* e_l1b  = (const float*)d_in[12];
  const float* e_l2W  = (const float*)d_in[13];
  const float* e_l2b  = (const float*)d_in[14];
  const float* e_l3W  = (const float*)d_in[15];
  const float* e_l3b  = (const float*)d_in[16];
  const float* e_n1b  = (const float*)d_in[17];
  const float* e_n1g  = (const float*)d_in[18];
  const float* e_n2b  = (const float*)d_in[19];
  const float* e_n2g  = (const float*)d_in[20];
  const float* g0W = (const float*)d_in[21];
  const float* g0b = (const float*)d_in[22];
  const float* g0lnb = (const float*)d_in[23];
  const float* g0lng = (const float*)d_in[24];
  const float* g1W = (const float*)d_in[25];
  const float* g1b = (const float*)d_in[26];
  const float* g1lnb = (const float*)d_in[27];
  const float* g1lng = (const float*)d_in[28];
  const float* g2W = (const float*)d_in[29];
  const float* g2b = (const float*)d_in[30];
  const float* g2lnb = (const float*)d_in[31];
  const float* g2lng = (const float*)d_in[32];
  const float* p_l1W = (const float*)d_in[33];
  const float* p_l1b = (const float*)d_in[34];
  const float* p_l2W = (const float*)d_in[35];
  const float* p_l2b = (const float*)d_in[36];
  const float* p_l3W = (const float*)d_in[37];
  const float* p_l3b = (const float*)d_in[38];
  const float* p_n1b = (const float*)d_in[39];
  const float* p_n1g = (const float*)d_in[40];
  const float* p_n2b = (const float*)d_in[41];
  const float* p_n2g = (const float*)d_in[42];
  const float* proj0 = (const float*)d_in[43];
  const int* edge_index   = (const int*)d_in[44];
  const int* action_edges = (const int*)d_in[45];
  const int* army_counts  = (const int*)d_in[46];

  const int N = in_sizes[0] / 128;
  const int E = in_sizes[44] / 2;
  const int A = in_sizes[45] / 2;
  const int* src = edge_index;
  const int* tgt = edge_index + E;

  // Workspace carve (256B aligned): deg, dis, xw, agg, hA, hB, ident
  char* wp = (char*)d_ws;
  auto carve = [&](size_t bytes) {
    void* r = (void*)wp;
    wp += (bytes + 255) & ~(size_t)255;
    return r;
  };
  float* deg   = (float*)carve((size_t)N * 4);
  float* dis   = (float*)carve((size_t)N * 4);
  float* xw    = (float*)carve((size_t)N * 64 * 4);
  float* agg   = (float*)carve((size_t)N * 64 * 4);
  float* hA    = (float*)carve((size_t)N * 64 * 4);
  float* hB    = (float*)carve((size_t)N * 64 * 4);
  float* ident = (float*)carve((size_t)N * 64 * 4);

  float* outP = (float*)d_out;
  float* outE = outP + N;
  float* outA = outE + A;

  const int TB = 256;
  int gN   = (N + TB - 1) / TB;
  int gE   = (E + TB - 1) / TB;
  int gNE  = (N * 64 + TB - 1) / TB;
  int gSc  = (E * 16 + TB - 1) / TB;
  int gGem = (N + 63) / 64;       // 4 waves x 16 rows
  int gPost = (N + 7) / 8;        // 8 waves x 1 node
  int gPlc = ((N + 15) / 16 + 3) / 4;
  int gEA  = ((A + 15) / 16 + 3) / 4;

  // Degrees and symmetric normalization
  init_deg_kernel<<<gN, TB, 0, stream>>>(deg, N);
  accum_deg_kernel<<<gE, TB, 0, stream>>>(tgt, deg, E);
  rsqrt_deg_kernel<<<gN, TB, 0, stream>>>(deg, dis, N);

  // ---- GCN layer 0 (F=128 -> 64, identity = x @ proj0) ----
  gemm_wmma_kernel<128, 64><<<gGem, 128, 0, stream>>>(x, g0W, xw, N);
  gemm_wmma_kernel<128, 64><<<gGem, 128, 0, stream>>>(x, proj0, ident, N);
  init_agg_kernel<<<gNE, TB, 0, stream>>>(xw, deg, g0b, agg, N);
  scatter_kernel<<<gSc, TB, 0, stream>>>(src, tgt, dis, xw, agg, E);
  gcn_post_kernel<<<gPost, TB, 0, stream>>>(agg, ident, g0lng, g0lnb, hA, N);

  // ---- GCN layer 1 (64 -> 64, identity = h) ----
  gemm_wmma_kernel<64, 64><<<gGem, 128, 0, stream>>>(hA, g1W, xw, N);
  init_agg_kernel<<<gNE, TB, 0, stream>>>(xw, deg, g1b, agg, N);
  scatter_kernel<<<gSc, TB, 0, stream>>>(src, tgt, dis, xw, agg, E);
  gcn_post_kernel<<<gPost, TB, 0, stream>>>(agg, hA, g1lng, g1lnb, hB, N);

  // ---- GCN layer 2 (64 -> 64, identity = h) ----
  gemm_wmma_kernel<64, 64><<<gGem, 128, 0, stream>>>(hB, g2W, xw, N);
  init_agg_kernel<<<gNE, TB, 0, stream>>>(xw, deg, g2b, agg, N);
  scatter_kernel<<<gSc, TB, 0, stream>>>(src, tgt, dis, xw, agg, E);
  gcn_post_kernel<<<gPost, TB, 0, stream>>>(agg, hB, g2lng, g2lnb, hA, N);

  // ---- Heads (emb = hA) ----
  place_mlp_kernel<<<gPlc, 128, 0, stream>>>(
      hA, p_l1W, p_l1b, p_n1g, p_n1b, p_l2W, p_l2b, p_n2g, p_n2b,
      p_l3W, p_l3b, outP, N);

  edge_army_kernel<<<gEA, 128, 0, stream>>>(
      hA, action_edges, army_counts,
      e_l1W, e_l1b, e_n1g, e_n1b, e_l2W, e_l2b, e_n2g, e_n2b, e_l3W, e_l3b,
      a_l1W, a_l1b, a_n1g, a_n1b, a_l2W, a_l2b, a_n2g, a_n2b, a_l3W, a_l3b,
      outE, outA, A, N);
}
